// StructuredInferenceLR_50062138802560
// MI455X (gfx1250) — compile-verified, matrix operand order checked by reading, not run on hardware
//
#include <hip/hip_runtime.h>
#include <hip/hip_bf16.h>

// ---------------------------------------------------------------------------
// StructuredInferenceLR for MI455X (gfx1250).
// - All GEMMs: v_wmma_f32_16x16x32_bf16 (bf16 A/B, f32 accumulate).
// - Single persistent kernel pipelines all 16 (dir,layer) LSTM scans with a
//   1-timestep skew via acquire/release progress flags (serial span ~T+7
//   instead of 16*T).
// - Embedding folded into layer-0 weights: xg0 = obs @ (Wih0@emb_w).T + bias.
// ---------------------------------------------------------------------------

typedef __bf16 bf16_t;
typedef __bf16 v16bf __attribute__((ext_vector_type(16)));
typedef float  v8f   __attribute__((ext_vector_type(8)));

#define WMMA_BF16(A, B, C) \
  __builtin_amdgcn_wmma_f32_16x16x32_bf16(false, (A), false, (B), (short)0, (C), false, false)

constexpr int Bsz = 32, Tsz = 2048, Osz = 128, Hsz = 256, Lsz = 8, Psz = 64;
constexpr int G4H = 4 * Hsz;          // 1024 gate width
constexpr int TB  = Tsz * Bsz;        // 65536 rows (time-major)

// K index inside a 16-bit A/B fragment (16x32 tile), per CDNA5 ISA 7.12.2.
__device__ __forceinline__ int frag_k(int lane, int e) {
  int v = e >> 1;
  return ((v >> 2) << 4) + ((lane >> 4) << 3) + ((v & 3) << 1) + (e & 1);
}
// C/D fragment (16x16 f32): element e -> row = (lane>>4)*8 + e, col = lane&15.

__device__ __forceinline__ float sigf(float x) { return 1.0f / (1.0f + __expf(-x)); }

// ---------------------------------------------------------------------------
// Small setup kernels
// ---------------------------------------------------------------------------
__global__ void k_f32_to_bf16(const float* __restrict__ in, bf16_t* __restrict__ out, int n) {
  int i = blockIdx.x * blockDim.x + threadIdx.x;
  if (i < n) out[i] = (bf16_t)in[i];
}

__global__ void k_zero_i32(int* __restrict__ p, int n) {
  int i = blockIdx.x * blockDim.x + threadIdx.x;
  if (i < n) p[i] = 0;
}

// obs (B,T,O) fp32 -> time-major bf16, forward and time-flipped copies.
__global__ void k_obs_prep(const float* __restrict__ obs,
                           bf16_t* __restrict__ obsF, bf16_t* __restrict__ obsR) {
  size_t i = (size_t)blockIdx.x * blockDim.x + threadIdx.x;
  if (i >= (size_t)Bsz * Tsz * Osz) return;
  int o = (int)(i % Osz);
  size_t r = i / Osz;
  int t = (int)(r % Tsz), b = (int)(r / Tsz);
  bf16_t v = (bf16_t)obs[i];
  obsF[((size_t)t * Bsz + b) * Osz + o] = v;
  obsR[((size_t)(Tsz - 1 - t) * Bsz + b) * Osz + o] = v;
}

// Weff = Wih0 @ emb_w  : (4H,H)@(H,O) -> (4H,O), bf16 out.
__global__ void k_fold_w(const float* __restrict__ wih0, const float* __restrict__ embw,
                         bf16_t* __restrict__ weff) {
  int i = blockIdx.x * blockDim.x + threadIdx.x;
  if (i >= G4H * Osz) return;
  int g = i / Osz, o = i % Osz;
  float s = 0.f;
  for (int h = 0; h < Hsz; ++h) s += wih0[g * Hsz + h] * embw[h * Osz + o];
  weff[i] = (bf16_t)s;
}

// biasC[l][g] = bih[l][g] + bhh[l][g]  (+ Wih0[g,:]@emb_b for l==0)
__global__ void k_bias_combine(const float* __restrict__ bih, const float* __restrict__ bhh,
                               const float* __restrict__ wih0, const float* __restrict__ embb,
                               float* __restrict__ biasC) {
  int i = blockIdx.x * blockDim.x + threadIdx.x;
  if (i >= Lsz * G4H) return;
  float v = bih[i] + bhh[i];
  if (i < G4H) {
    float s = 0.f;
    for (int h = 0; h < Hsz; ++h) s += wih0[i * Hsz + h] * embb[h];
    v += s;
  }
  biasC[i] = v;
}

// ---------------------------------------------------------------------------
// Layer-0 input-gate GEMM: xg0 = obs_tm @ Weff.T + biasC[0]  (fp32, time-major)
// M = T*B, N = 1024, K = 128. 4 M-tiles per wave (B fragment reused).
// ---------------------------------------------------------------------------
__global__ __launch_bounds__(256) void k_gate_gemm(const bf16_t* __restrict__ hin, int K,
                                                   const bf16_t* __restrict__ wih,
                                                   const float* __restrict__ biasC,
                                                   float* __restrict__ xg) {
  const int lane = threadIdx.x & 31;
  const int wave = threadIdx.x >> 5;
  const int nN   = G4H / 16;                  // 64
  long job = (long)blockIdx.x * 8 + wave;
  int mg = (int)(job / nN), nt = (int)(job % nN);
  int m0 = mg * 64, n0 = nt * 16;
  const int nn  = lane & 15;
  const int mlo = (lane >> 4) << 3;

  v8f acc[4] = {};
  for (int kc = 0; kc < K / 32; ++kc) {
    v16bf b;
#pragma unroll
    for (int e = 0; e < 16; ++e)
      b[e] = wih[(size_t)(n0 + nn) * K + kc * 32 + frag_k(lane, e)];
#pragma unroll
    for (int i = 0; i < 4; ++i) {
      const bf16_t* arow = hin + (size_t)(m0 + i * 16 + nn) * K;
      v16bf a;
#pragma unroll
      for (int e = 0; e < 16; ++e)
        a[e] = arow[kc * 32 + frag_k(lane, e)];
      acc[i] = WMMA_BF16(a, b, acc[i]);
    }
  }
  float bsum = biasC[n0 + nn];
#pragma unroll
  for (int i = 0; i < 4; ++i)
#pragma unroll
    for (int e = 0; e < 8; ++e)
      xg[(size_t)(m0 + i * 16 + mlo + e) * G4H + n0 + nn] = acc[i][e] + bsum;
}

// ---------------------------------------------------------------------------
// Fused persistent bidirectional LSTM pipeline.
// grid = 32 blocks: bid = d*16 + l*2 + half   (d: dir, l: layer, half: 16 rows)
// Layer 0 streams precomputed xg0; layers 1..7 compute input gates on the fly
// from the producer layer's h (published via release atomic, consumed with
// acquire + s_sleep spin). Cell state + whh/wr fragments stay in registers.
// ---------------------------------------------------------------------------
__global__ __launch_bounds__(512) void k_lstm_pipe(
    const float* __restrict__ xgF, const float* __restrict__ xgR,
    const bf16_t* __restrict__ wihF, const bf16_t* __restrict__ wihB,
    const bf16_t* __restrict__ whhF, const bf16_t* __restrict__ whhB,
    const bf16_t* __restrict__ wrF,  const bf16_t* __restrict__ wrB,
    const float*  __restrict__ biasF, const float* __restrict__ biasB,
    bf16_t* __restrict__ hbuf, int* __restrict__ prog) {
  __shared__ bf16_t s_wih[G4H * Psz];   // 128 KB (layers >= 1)
  __shared__ bf16_t s_ot[16 * Hsz];     //   8 KB  o*tanh(c)
  __shared__ bf16_t s_h[16 * Psz];      //   2 KB  projected hidden

  const int bid  = blockIdx.x;
  const int d    = bid >> 4;
  const int l    = (bid >> 1) & 7;
  const int half = bid & 1;
  const int lane = threadIdx.x & 31;
  const int wave = threadIdx.x >> 5;
  const int b0   = half * 16;
  const int nn   = lane & 15;
  const int mlo  = (lane >> 4) << 3;
  const int j0   = wave * 16;

  const float*  xg   = d ? xgR : xgF;
  const bf16_t* wih  = (d ? wihB : wihF) + (size_t)G4H * Osz + (size_t)(l - 1) * G4H * Psz;
  const bf16_t* whh  = (d ? whhB : whhF) + (size_t)l * G4H * Psz;
  const bf16_t* wr   = (d ? wrB  : wrF ) + (size_t)l * Psz * Hsz;
  const float*  bias = (d ? biasB : biasF) + l * G4H;
  const bf16_t* hin  = hbuf + (size_t)(d * Lsz + (l - 1)) * TB * Psz;  // l>=1 only
  bf16_t* hout = hbuf + (size_t)(d * Lsz + l) * TB * Psz;
  int* myprog = prog + (d * Lsz + l) * 2 + half;
  int* inprog = prog + (d * Lsz + l - 1) * 2 + half;                   // l>=1 only

  // Stage this layer's Wih in LDS (layers >= 1).
  if (l > 0)
    for (int i = threadIdx.x; i < G4H * Psz; i += blockDim.x) s_wih[i] = wih[i];

  // Register-resident recurrent weights.
  v16bf bw[2][4];                       // whh fragments [k-chunk][gate]
#pragma unroll
  for (int kc = 0; kc < 2; ++kc)
#pragma unroll
    for (int g = 0; g < 4; ++g)
#pragma unroll
      for (int e = 0; e < 16; ++e)
        bw[kc][g][e] = whh[(size_t)(g * Hsz + j0 + nn) * Psz + kc * 32 + frag_k(lane, e)];

  v16bf bwr[8];                         // wr fragments (projection waves)
  if (wave < 4) {
#pragma unroll
    for (int kc = 0; kc < 8; ++kc)
#pragma unroll
      for (int e = 0; e < 16; ++e)
        bwr[kc][e] = wr[(size_t)(wave * 16 + nn) * Hsz + kc * 32 + frag_k(lane, e)];
  }

  // Per-gate bias (broadcast over rows), layers >= 1.
  float bb[4];
#pragma unroll
  for (int g = 0; g < 4; ++g) bb[g] = bias[g * Hsz + j0 + nn];

  for (int i = threadIdx.x; i < 16 * Psz; i += blockDim.x) s_h[i] = (bf16_t)0.0f;
  __syncthreads();

  v8f cfrag = {};                       // cell state for this wave's 16 units
  const bool publish = (l != Lsz - 1);

  for (int t = 0; t < Tsz; ++t) {
    v8f g0, g1, g2, g3;
    if (l == 0) {
      // Stream precomputed xg0 (immutable -> prefetch next slab is safe).
      if (t + 1 < Tsz) {
        const char* nx = (const char*)(xg + ((size_t)(t + 1) * Bsz + b0) * G4H);
        __builtin_prefetch(nx + (size_t)threadIdx.x * 128, 0, 3);
      }
      const float* xrow = xg + ((size_t)t * Bsz + b0) * G4H;
#pragma unroll
      for (int e = 0; e < 8; ++e) {
        const float* xr = xrow + (size_t)(mlo + e) * G4H;
        g0[e] = xr[0 * Hsz + j0 + nn];
        g1[e] = xr[1 * Hsz + j0 + nn];
        g2[e] = xr[2 * Hsz + j0 + nn];
        g3[e] = xr[3 * Hsz + j0 + nn];
      }
    } else {
      // Wait for producer layer to publish h(t); acquire invalidates WGP$.
      while (__hip_atomic_load(inprog, __ATOMIC_ACQUIRE, __HIP_MEMORY_SCOPE_AGENT) < t + 1)
        __builtin_amdgcn_s_sleep(2);
#pragma unroll
      for (int e = 0; e < 8; ++e) { g0[e] = bb[0]; g1[e] = bb[1]; g2[e] = bb[2]; g3[e] = bb[3]; }
      // Input-gate GEMM on the fly: h_in(t) @ Wih.T
      const bf16_t* hrow = hin + (size_t)(t * Bsz + b0 + nn) * Psz;
#pragma unroll
      for (int kc = 0; kc < 2; ++kc) {
        v16bf a, w0, w1, w2, w3;
#pragma unroll
        for (int e = 0; e < 16; ++e) {
          int k = kc * 32 + frag_k(lane, e);
          a[e]  = hrow[k];
          w0[e] = s_wih[(0 * Hsz + j0 + nn) * Psz + k];
          w1[e] = s_wih[(1 * Hsz + j0 + nn) * Psz + k];
          w2[e] = s_wih[(2 * Hsz + j0 + nn) * Psz + k];
          w3[e] = s_wih[(3 * Hsz + j0 + nn) * Psz + k];
        }
        g0 = WMMA_BF16(a, w0, g0);
        g1 = WMMA_BF16(a, w1, g1);
        g2 = WMMA_BF16(a, w2, g2);
        g3 = WMMA_BF16(a, w3, g3);
      }
    }

    // Recurrent contribution: h(t-1) @ whh.T
#pragma unroll
    for (int kc = 0; kc < 2; ++kc) {
      v16bf a;
#pragma unroll
      for (int e = 0; e < 16; ++e)
        a[e] = s_h[nn * Psz + kc * 32 + frag_k(lane, e)];
      g0 = WMMA_BF16(a, bw[kc][0], g0);
      g1 = WMMA_BF16(a, bw[kc][1], g1);
      g2 = WMMA_BF16(a, bw[kc][2], g2);
      g3 = WMMA_BF16(a, bw[kc][3], g3);
    }

    // Gate nonlinearity + in-register cell update (i,f,g,o order).
#pragma unroll
    for (int e = 0; e < 8; ++e) {
      float iv = sigf(g0[e]);
      float fv = sigf(g1[e]);
      float gv = tanhf(g2[e]);
      float ov = sigf(g3[e]);
      float c  = fv * cfrag[e] + iv * gv;
      cfrag[e] = c;
      s_ot[(mlo + e) * Hsz + j0 + nn] = (bf16_t)(ov * tanhf(c));
    }
    __syncthreads();

    // Projection (waves 0..3): h = ot @ wr.T
    if (wave < 4) {
      v8f acc = {};
#pragma unroll
      for (int kc = 0; kc < 8; ++kc) {
        v16bf a;
#pragma unroll
        for (int e = 0; e < 16; ++e)
          a[e] = s_ot[nn * Hsz + kc * 32 + frag_k(lane, e)];
        acc = WMMA_BF16(a, bwr[kc], acc);
      }
      const int p0 = wave * 16;
#pragma unroll
      for (int e = 0; e < 8; ++e) {
        int m = mlo + e;
        bf16_t hv = (bf16_t)acc[e];
        s_h[m * Psz + p0 + nn] = hv;
        hout[((size_t)t * Bsz + b0 + m) * Psz + p0 + nn] = hv;
      }
    }
    if (publish) __threadfence();       // make h(t) visible at device scope
    __syncthreads();
    if (publish && threadIdx.x == 0)
      __hip_atomic_store(myprog, t + 1, __ATOMIC_RELEASE, __HIP_MEMORY_SCOPE_AGENT);
  }
}

// ---------------------------------------------------------------------------
// Latent scan: one persistent workgroup (16 waves), WMMA 32x64 @ 64x64 steps.
// ---------------------------------------------------------------------------
__global__ __launch_bounds__(512) void k_latent(const bf16_t* __restrict__ outl,
                                                const bf16_t* __restrict__ hbwd,
                                                const float* __restrict__ eps,
                                                const bf16_t* __restrict__ wc,
                                                const float* __restrict__ bc,
                                                const bf16_t* __restrict__ wmu,
                                                const float* __restrict__ bmu,
                                                const bf16_t* __restrict__ wlv,
                                                const float* __restrict__ blv,
                                                float* __restrict__ mus,
                                                float* __restrict__ lvs,
                                                float* __restrict__ zs) {
  __shared__ bf16_t s_z[Bsz * Psz];
  __shared__ bf16_t s_hm[Bsz * Psz];
  __shared__ float  s_mu[Bsz * Psz];
  __shared__ float  s_lv[Bsz * Psz];

  const int lane = threadIdx.x & 31;
  const int wave = threadIdx.x >> 5;
  const int nn   = lane & 15;
  const int mlo  = (lane >> 4) << 3;

  const int ntw = wave & 3;
  v16bf wcf[2], wsf[2];
  const bf16_t* wsel = (wave < 8) ? wmu : wlv;
#pragma unroll
  for (int kc = 0; kc < 2; ++kc)
#pragma unroll
    for (int e = 0; e < 16; ++e) {
      int k = kc * 32 + frag_k(lane, e);
      wcf[kc][e] = wc[(size_t)(ntw * 16 + nn) * Psz + k];
      wsf[kc][e] = wsel[(size_t)(ntw * 16 + nn) * Psz + k];
    }

  for (int i = threadIdx.x; i < Bsz * Psz; i += blockDim.x) s_z[i] = (bf16_t)0.0f;
  __syncthreads();

  for (int t = 0; t < Tsz; ++t) {
    if (wave < 8) {
      int mt = wave >> 2;
      v8f acc = {};
#pragma unroll
      for (int kc = 0; kc < 2; ++kc) {
        v16bf a;
#pragma unroll
        for (int e = 0; e < 16; ++e)
          a[e] = s_z[(mt * 16 + nn) * Psz + kc * 32 + frag_k(lane, e)];
        acc = WMMA_BF16(a, wcf[kc], acc);
      }
#pragma unroll
      for (int e = 0; e < 8; ++e) {
        int m = mt * 16 + mlo + e;
        int p = ntw * 16 + nn;
        float ls = (float)outl[((size_t)t * Bsz + m) * Psz + p] +
                   (float)hbwd[((size_t)(Tsz - 1 - t) * Bsz + m) * Psz + p];
        float h = (tanhf(acc[e] + bc[p]) + ls) * (1.0f / 3.0f);
        s_hm[m * Psz + p] = (bf16_t)h;
      }
    }
    __syncthreads();
    {
      int mt = (wave >> 2) & 1;
      const float* bias = (wave < 8) ? bmu : blv;
      v8f acc = {};
#pragma unroll
      for (int kc = 0; kc < 2; ++kc) {
        v16bf a;
#pragma unroll
        for (int e = 0; e < 16; ++e)
          a[e] = s_hm[(mt * 16 + nn) * Psz + kc * 32 + frag_k(lane, e)];
        acc = WMMA_BF16(a, wsf[kc], acc);
      }
#pragma unroll
      for (int e = 0; e < 8; ++e) {
        int m = mt * 16 + mlo + e;
        int p = ntw * 16 + nn;
        float v = acc[e] + bias[p];
        if (wave < 8) { s_mu[m * Psz + p] = v; mus[((size_t)m * Tsz + t) * Psz + p] = v; }
        else          { s_lv[m * Psz + p] = v; lvs[((size_t)m * Tsz + t) * Psz + p] = v; }
      }
    }
    __syncthreads();
    for (int i = threadIdx.x; i < Bsz * Psz; i += blockDim.x) {
      int m = i / Psz, p = i % Psz;
      float z = s_mu[i] + eps[((size_t)m * Tsz + t) * Psz + p] * __expf(0.5f * s_lv[i]);
      zs[((size_t)m * Tsz + t) * Psz + p] = z;
      s_z[i] = (bf16_t)z;
    }
    __syncthreads();
  }
}

// ---------------------------------------------------------------------------
// Host orchestration
// ---------------------------------------------------------------------------
extern "C" void kernel_launch(void* const* d_in, const int* in_sizes, int n_in,
                              void* d_out, int out_size, void* d_ws, size_t ws_size,
                              hipStream_t stream) {
  (void)in_sizes; (void)n_in; (void)out_size; (void)ws_size;

  const float* obs   = (const float*)d_in[0];
  const float* eps   = (const float*)d_in[1];
  const float* embw  = (const float*)d_in[2];
  const float* embb  = (const float*)d_in[3];
  const float* wih0[2] = {(const float*)d_in[4],  (const float*)d_in[10]};
  const float* wihr[2] = {(const float*)d_in[5],  (const float*)d_in[11]};
  const float* whh[2]  = {(const float*)d_in[6],  (const float*)d_in[12]};
  const float* bih[2]  = {(const float*)d_in[7],  (const float*)d_in[13]};
  const float* bhh[2]  = {(const float*)d_in[8],  (const float*)d_in[14]};
  const float* wr[2]   = {(const float*)d_in[9],  (const float*)d_in[15]};
  const float* wc  = (const float*)d_in[16]; const float* bc  = (const float*)d_in[17];
  const float* wmu = (const float*)d_in[18]; const float* bmu = (const float*)d_in[19];
  const float* wlv = (const float*)d_in[20]; const float* blv = (const float*)d_in[21];

  size_t off = 0;
  auto alloc = [&](size_t bytes) -> void* {
    void* p = (void*)((char*)d_ws + off);
    off += (bytes + 255) & ~(size_t)255;
    return p;
  };
  auto conv = [&](const float* src, bf16_t* dst, int n) {
    k_f32_to_bf16<<<(n + 255) / 256, 256, 0, stream>>>(src, dst, n);
  };

  // bf16 weights: wihAll[d] = [Weff (1024x128)] [wih layers 1..7 (1024x64 each)]
  constexpr size_t WIH_ALL = (size_t)G4H * Osz + (size_t)(Lsz - 1) * G4H * Psz;
  bf16_t *c_wih[2], *c_whh[2], *c_wr[2];
  float* c_bias[2];
  for (int d = 0; d < 2; ++d) {
    c_wih[d]  = (bf16_t*)alloc(WIH_ALL * 2);
    c_whh[d]  = (bf16_t*)alloc((size_t)Lsz * G4H * Psz * 2);
    c_wr[d]   = (bf16_t*)alloc((size_t)Lsz * Psz * Hsz * 2);
    c_bias[d] = (float*)alloc((size_t)Lsz * G4H * 4);
  }
  bf16_t* c_wc  = (bf16_t*)alloc((size_t)Psz * Psz * 2);
  bf16_t* c_wmu = (bf16_t*)alloc((size_t)Psz * Psz * 2);
  bf16_t* c_wlv = (bf16_t*)alloc((size_t)Psz * Psz * 2);

  // activations
  bf16_t* obsF = (bf16_t*)alloc((size_t)TB * Osz * 2);
  bf16_t* obsR = (bf16_t*)alloc((size_t)TB * Osz * 2);
  bf16_t* hbuf = (bf16_t*)alloc((size_t)2 * Lsz * TB * Psz * 2);  // 134 MB
  int*    prog = (int*)alloc(32 * 4);
  float*  xgF  = (float*)alloc((size_t)TB * G4H * 4);             // 268 MB
  float*  xgR  = (float*)alloc((size_t)TB * G4H * 4);             // 268 MB

  // setup: conversions, folds, combined biases, obs prep, progress reset
  for (int d = 0; d < 2; ++d) {
    k_fold_w<<<(G4H * Osz + 255) / 256, 256, 0, stream>>>(wih0[d], embw, c_wih[d]);
    conv(wihr[d], c_wih[d] + (size_t)G4H * Osz, (Lsz - 1) * G4H * Psz);
    conv(whh[d], c_whh[d], Lsz * G4H * Psz);
    conv(wr[d],  c_wr[d],  Lsz * Psz * Hsz);
    k_bias_combine<<<(Lsz * G4H + 255) / 256, 256, 0, stream>>>(bih[d], bhh[d], wih0[d],
                                                                embb, c_bias[d]);
  }
  conv(wc, c_wc, Psz * Psz); conv(wmu, c_wmu, Psz * Psz); conv(wlv, c_wlv, Psz * Psz);
  {
    size_t n = (size_t)Bsz * Tsz * Osz;
    k_obs_prep<<<(int)((n + 255) / 256), 256, 0, stream>>>(obs, obsF, obsR);
  }
  k_zero_i32<<<1, 32, 0, stream>>>(prog, 32);

  // layer-0 xg precompute (big parallel GEMM, K=128, whole GPU)
  {
    long jobs = (long)(TB / 64) * (G4H / 16);
    k_gate_gemm<<<(int)(jobs / 8), 256, 0, stream>>>(obsF, Osz, c_wih[0], c_bias[0], xgF);
    k_gate_gemm<<<(int)(jobs / 8), 256, 0, stream>>>(obsR, Osz, c_wih[1], c_bias[1], xgR);
  }

  // fused persistent pipeline: 2 dirs x 8 layers x 2 batch-halves = 32 blocks
  k_lstm_pipe<<<32, 512, 0, stream>>>(xgF, xgR,
                                      c_wih[0], c_wih[1],
                                      c_whh[0], c_whh[1],
                                      c_wr[0],  c_wr[1],
                                      c_bias[0], c_bias[1],
                                      hbuf, prog);

  // latent scan (final layer outputs: (d*8+7))
  float* mus = (float*)d_out;
  float* lvs = mus + (size_t)Bsz * Tsz * Psz;
  float* zs  = lvs + (size_t)Bsz * Tsz * Psz;
  k_latent<<<1, 512, 0, stream>>>(hbuf + (size_t)(0 * Lsz + 7) * TB * Psz,
                                  hbuf + (size_t)(1 * Lsz + 7) * TB * Psz,
                                  eps, c_wc, bc, c_wmu, bmu, c_wlv, blv,
                                  mus, lvs, zs);
}
